// VanillaRNN_34402688041538
// MI455X (gfx1250) — compile-verified, matrix operand order checked by reading, not run on hardware
//
#include <hip/hip_runtime.h>
#include <hip/hip_bf16.h>

typedef __bf16 v16b __attribute__((ext_vector_type(16)));
typedef __bf16 v8b  __attribute__((ext_vector_type(8)));
typedef float  v8f  __attribute__((ext_vector_type(8)));

#define RNN_B 16384
#define RNN_D 512
#define RNN_H 1024
#define RNN_O 10

#define BT    64        // batch rows per workgroup
#define NW    16        // waves per workgroup (512 threads)
#define LSTR  1032      // padded LDS row stride in bf16 elems (2064 B, rotates banks)

union Frag { v16b f; v8b h[2]; };

// fast tanh via hardware transcendentals: tanh(x) = 1 - 2/(2^(2*log2e*x)+1)
// large +x: e->inf -> 1 ; large -x: e->0 -> -1 ; ~6 VALU ops vs ~25 for ocml tanhf
__device__ __forceinline__ float tanh_fast(float x) {
  const float e = __builtin_amdgcn_exp2f(x * 2.8853900817779268f);
  return 1.0f - 2.0f * __builtin_amdgcn_rcpf(e + 1.0f);
}

// ---------------------------------------------------------------- f32 -> bf16
__global__ void cvt_bf16(const float* __restrict__ src, __bf16* __restrict__ dst, int n) {
  int i = blockIdx.x * blockDim.x + threadIdx.x;
  if (i < n) dst[i] = (__bf16)src[i];
}

// ---------------------------------------------------------------------------
// xwb_frag = x @ Whx^T + (Whx_b + Whh_b), stored in C-fragment order:
//   offset = ((tm*64 + tn)*32 + lane)*8 floats,  tm = row/16, tn = col/16
// so rnn_steps can re-load accumulators with two b128 loads per tile.
// ---------------------------------------------------------------------------
__global__ __launch_bounds__(512)
void xw_gemm(const float* __restrict__ x,   const float* __restrict__ whx,
             const float* __restrict__ bhx, const float* __restrict__ bhh,
             float* __restrict__ xwb_frag)
{
  const int lane = threadIdx.x & 31;
  const int wv   = threadIdx.x >> 5;          // 0..15
  const int rowb = blockIdx.x * BT;           // batch row base
  const int colb = wv * 64;                   // this wave's 64 output cols
  const int lrow = lane & 15;
  const int khi  = (lane >> 4) << 3;          // 0 (lanes 0-15) or 8 (lanes 16-31)

  v8f acc[4][4];
#pragma unroll
  for (int nt = 0; nt < 4; ++nt) {
    const int c = colb + nt * 16 + lrow;
    const float bsum = bhx[c] + bhh[c];
#pragma unroll
    for (int mt = 0; mt < 4; ++mt)
#pragma unroll
      for (int v = 0; v < 8; ++v) acc[mt][nt][v] = bsum;
  }

  for (int kb = 0; kb < RNN_D / 32; ++kb) {
    const int k0 = kb * 32 + khi;
    Frag af[4], bf[4];
#pragma unroll
    for (int mt = 0; mt < 4; ++mt) {
      const float* ap = x + (size_t)(rowb + mt * 16 + lrow) * RNN_D + k0;
#pragma unroll
      for (int i = 0; i < 8; ++i) { af[mt].f[i] = (__bf16)ap[i]; af[mt].f[8 + i] = (__bf16)ap[16 + i]; }
    }
#pragma unroll
    for (int nt = 0; nt < 4; ++nt) {
      const float* bp = whx + (size_t)(colb + nt * 16 + lrow) * RNN_D + k0;
#pragma unroll
      for (int i = 0; i < 8; ++i) { bf[nt].f[i] = (__bf16)bp[i]; bf[nt].f[8 + i] = (__bf16)bp[16 + i]; }
    }
#pragma unroll
    for (int mt = 0; mt < 4; ++mt)
#pragma unroll
      for (int nt = 0; nt < 4; ++nt)
        acc[mt][nt] = __builtin_amdgcn_wmma_f32_16x16x32_bf16(
            false, af[mt].f, false, bf[nt].f, (short)0, acc[mt][nt], false, false);
  }

  // store in fragment order: one v8f (2 x b128) per tile
#pragma unroll
  for (int mt = 0; mt < 4; ++mt) {
    const size_t tm = (size_t)blockIdx.x * 4 + mt;
#pragma unroll
    for (int nt = 0; nt < 4; ++nt) {
      const size_t tn = (size_t)wv * 4 + nt;
      *(v8f*)(xwb_frag + ((tm * 64 + tn) * 32 + lane) * 8) = acc[mt][nt];
    }
  }
}

// ---------------------------------------------------------------------------
// Persistent recurrence: per-WG 64-row h tile lives in LDS for all S steps.
//   h = tanh(xWb + h @ Whh^T)     (bias pre-folded into xWb)
// ---------------------------------------------------------------------------
__global__ __launch_bounds__(512)
void rnn_steps(const float* __restrict__ xwb_frag, const __bf16* __restrict__ whh_bf,
               const int* __restrict__ seqlen, __bf16* __restrict__ hfin)
{
  __shared__ __bf16 hbuf[BT * LSTR];          // 64 x 1032 bf16 = 132,096 B of LDS

  const int lane = threadIdx.x & 31;
  const int wv   = threadIdx.x >> 5;
  const int rowb = blockIdx.x * BT;
  const int colb = wv * 64;
  const int lrow = lane & 15;
  const int khi  = (lane >> 4) << 3;
  const int rhi  = (lane >> 4) << 3;
  const int nS   = seqlen[0];

  for (int i = threadIdx.x; i < BT * LSTR; i += blockDim.x) hbuf[i] = (__bf16)0.0f;
  __syncthreads();

  v8f acc[4][4];

  for (int s = 0; s < nS; ++s) {
    // accumulator init from loop-invariant, fragment-ordered xWb (L2-resident)
#pragma unroll
    for (int mt = 0; mt < 4; ++mt) {
      const size_t tm = (size_t)blockIdx.x * 4 + mt;
#pragma unroll
      for (int nt = 0; nt < 4; ++nt) {
        const size_t tn = (size_t)wv * 4 + nt;
        acc[mt][nt] = *(const v8f*)(xwb_frag + ((tm * 64 + tn) * 32 + lane) * 8);
      }
    }

    // K loop over H=1024 in steps of 32
    for (int kb = 0; kb < RNN_H / 32; ++kb) {
      const int k0 = kb * 32 + khi;
      Frag af[4], bf[4];
#pragma unroll
      for (int mt = 0; mt < 4; ++mt) {
        const __bf16* ap = hbuf + (mt * 16 + lrow) * LSTR + k0;
        af[mt].h[0] = *(const v8b*)ap;
        af[mt].h[1] = *(const v8b*)(ap + 16);
      }
#pragma unroll
      for (int nt = 0; nt < 4; ++nt) {
        const __bf16* bp = whh_bf + (size_t)(colb + nt * 16 + lrow) * RNN_H + k0;
        bf[nt].h[0] = *(const v8b*)bp;
        bf[nt].h[1] = *(const v8b*)(bp + 16);
      }
#pragma unroll
      for (int mt = 0; mt < 4; ++mt)
#pragma unroll
        for (int nt = 0; nt < 4; ++nt)
          acc[mt][nt] = __builtin_amdgcn_wmma_f32_16x16x32_bf16(
              false, af[mt].f, false, bf[nt].f, (short)0, acc[mt][nt], false, false);
    }

    __syncthreads();   // all waves done reading hbuf for this step

#pragma unroll
    for (int mt = 0; mt < 4; ++mt)
#pragma unroll
      for (int nt = 0; nt < 4; ++nt)
#pragma unroll
        for (int v = 0; v < 8; ++v) {
          const int r = mt * 16 + v + rhi;
          const int c = colb + nt * 16 + lrow;
          hbuf[r * LSTR + c] = (__bf16)tanh_fast(acc[mt][nt][v]);
        }

    __syncthreads();   // hbuf fully updated before next step's reads
  }

  // coalesced final writeback: LDS -> global, 16B per thread per iter
  for (int i = threadIdx.x; i < BT * (RNN_H / 8); i += blockDim.x) {
    const int r  = i >> 7;          // 128 x 8-elem chunks per row
    const int c8 = (i & 127) * 8;
    *(v8b*)(hfin + (size_t)(rowb + r) * RNN_H + c8) = *(const v8b*)(hbuf + r * LSTR + c8);
  }
}

// ---------------------------------------------------------------------------
// out[B][10] = softmax(h @ Wph^T + b)   — one wave per batch row
// ---------------------------------------------------------------------------
__global__ __launch_bounds__(256)
void out_softmax(const __bf16* __restrict__ h, const float* __restrict__ wph,
                 const float* __restrict__ bph, float* __restrict__ out)
{
  const int lane = threadIdx.x & 31;
  const int row  = blockIdx.x * 8 + (threadIdx.x >> 5);
  const __bf16* hr = h + (size_t)row * RNN_H;

  float p[RNN_O];
#pragma unroll
  for (int o = 0; o < RNN_O; ++o) p[o] = 0.0f;

  for (int c = lane; c < RNN_H; c += 32) {
    const float hv = (float)hr[c];
#pragma unroll
    for (int o = 0; o < RNN_O; ++o) p[o] += hv * wph[o * RNN_H + c];
  }
#pragma unroll
  for (int o = 0; o < RNN_O; ++o)
    for (int off = 16; off; off >>= 1) p[o] += __shfl_xor(p[o], off, 32);

  if (lane == 0) {
    float l[RNN_O], m = -3.4e38f, ssum = 0.0f;
#pragma unroll
    for (int o = 0; o < RNN_O; ++o) { l[o] = p[o] + bph[o]; m = fmaxf(m, l[o]); }
#pragma unroll
    for (int o = 0; o < RNN_O; ++o) { l[o] = expf(l[o] - m); ssum += l[o]; }
    const float inv = 1.0f / ssum;
#pragma unroll
    for (int o = 0; o < RNN_O; ++o) out[(size_t)row * RNN_O + o] = l[o] * inv;
  }
}

// ---------------------------------------------------------------------------
extern "C" void kernel_launch(void* const* d_in, const int* in_sizes, int n_in,
                              void* d_out, int out_size, void* d_ws, size_t ws_size,
                              hipStream_t stream) {
  const float* x    = (const float*)d_in[0];
  const float* whx  = (const float*)d_in[1];
  const float* bhx  = (const float*)d_in[2];
  const float* whh  = (const float*)d_in[3];
  const float* bhh  = (const float*)d_in[4];
  const float* wph  = (const float*)d_in[5];
  const float* bph  = (const float*)d_in[6];
  const int* seqlen = (const int*)d_in[7];
  float* out        = (float*)d_out;

  // workspace layout
  char* ws = (char*)d_ws;
  __bf16* whh_bf = (__bf16*)ws;                                   //  2 MB
  float*  xwb    = (float*)(ws + (size_t)(2u << 20));             // 64 MB (fragment order)
  __bf16* hfin   = (__bf16*)(ws + (size_t)(2u << 20) + (size_t)(64u << 20)); // 32 MB

  cvt_bf16<<<(RNN_H * RNN_H + 255) / 256, 256, 0, stream>>>(whh, whh_bf, RNN_H * RNN_H);
  xw_gemm<<<RNN_B / BT, NW * 32, 0, stream>>>(x, whx, bhx, bhh, xwb);
  rnn_steps<<<RNN_B / BT, NW * 32, 0, stream>>>(xwb, whh_bf, seqlen, hfin);
  out_softmax<<<RNN_B / 8, 256, 0, stream>>>(hfin, wph, bph, out);
}
